// HMModel_37830071943171
// MI455X (gfx1250) — matrix operand order repaired and checked by reading, not compiled, not run on hardware
//
#include <hip/hip_runtime.h>
#include <stdint.h>

typedef float  v4f   __attribute__((ext_vector_type(4)));
typedef float  v8f   __attribute__((ext_vector_type(8)));
typedef __bf16 v16bf __attribute__((ext_vector_type(16)));

constexpr int B  = 16, H = 50, V = 100000, D = 512;
constexpr int HP = 64;     // H padded to 4 WMMA M-tiles
constexpr int NT = 256;    // v-columns per block (8 waves x 2 subtiles x 16)
constexpr int KC = 64;     // K chunk in bf16 elements (2 WMMA k-steps)
constexpr int LDA = 36;    // LDS row stride in dwords (32 data + 4 pad, 16B aligned)
constexpr int LDB = 36;

__device__ __forceinline__ uint32_t f32_to_bf16(float f) {
  union { float f; uint32_t u; } v; v.f = f;
  return (v.u + 0x7fffu + ((v.u >> 16) & 1u)) >> 16;   // round-to-nearest-even
}

// ---------------------------------------------------------------------------
// Kernel 1: L2-normalize every emb row, write bf16 table (102.4 MB, fits L2).
// ---------------------------------------------------------------------------
__global__ __launch_bounds__(128) void k_normalize(const float* __restrict__ emb,
                                                   uint32_t* __restrict__ embN) {
  const int v = blockIdx.x;
  const int t = threadIdx.x;
  const v4f x = *(const v4f*)(emb + (size_t)v * D + t * 4);
  float ss = x.x * x.x + x.y * x.y + x.z * x.z + x.w * x.w;
  #pragma unroll
  for (int off = 16; off > 0; off >>= 1) ss += __shfl_xor(ss, off, 32);
  __shared__ float sp[4];
  if ((t & 31) == 0) sp[t >> 5] = ss;
  __syncthreads();
  const float inv = 1.0f / fmaxf(sqrtf(sp[0] + sp[1] + sp[2] + sp[3]), 1e-12f);
  const uint32_t lo = f32_to_bf16(x.x * inv) | (f32_to_bf16(x.y * inv) << 16);
  const uint32_t hi = f32_to_bf16(x.z * inv) | (f32_to_bf16(x.w * inv) << 16);
  *(uint64_t*)(embN + (size_t)v * (D / 2) + t * 2) = (uint64_t)lo | ((uint64_t)hi << 32);
}

// ---------------------------------------------------------------------------
// Kernel 2: gather normalized history rows into padded A [B,64,512] bf16.
// ---------------------------------------------------------------------------
__global__ __launch_bounds__(256) void k_gather_hist(const int* __restrict__ hist,
                                                     const uint32_t* __restrict__ embN,
                                                     uint32_t* __restrict__ Aws) {
  const int b  = blockIdx.x >> 6;
  const int hh = blockIdx.x & 63;
  const int t  = threadIdx.x;
  uint32_t val = 0;
  if (hh < H) {
    const int a = hist[b * H + hh];
    val = embN[(size_t)a * (D / 2) + t];
  }
  Aws[((size_t)(b * HP + hh)) * (D / 2) + t] = val;
}

// ---------------------------------------------------------------------------
// Epilogue helpers
// ---------------------------------------------------------------------------
__device__ __forceinline__ void reduce_cols(const v8f& a0, const v8f& a1,
                                            const v8f& a2, const v8f& a3,
                                            int hi, float& best, int& bh) {
  best = -1e30f; bh = 0;
  #pragma unroll
  for (int t = 0; t < 4; ++t) {
    const v8f acc = (t == 0) ? a0 : (t == 1) ? a1 : (t == 2) ? a2 : a3;
    #pragma unroll
    for (int r = 0; r < 8; ++r) {
      const int h = 16 * t + r + 8 * hi;
      if (h < H) {
        const float val = acc[r];
        if (val > best) { best = val; bh = h; }
      }
    }
  }
  const float oBest = __shfl_xor(best, 16, 32);
  const int   oBh   = __shfl_xor(bh,   16, 32);
  if (oBest > best || (oBest == best && oBh < bh)) { best = oBest; bh = oBh; }
}

__device__ __forceinline__ float mlp_head(float best, float wk, float lk,
                                          const float* __restrict__ w1, const float* __restrict__ b1,
                                          const float* __restrict__ w2, const float* __restrict__ b2,
                                          const float* __restrict__ w3, const float* __restrict__ b3) {
  const float p     = fminf(fmaxf(best, 0.001f), 0.999f);
  const float logit = -logf(1.0f / p - 1.0f);
  float h1[32];
  #pragma unroll
  for (int o = 0; o < 32; ++o) {
    const float s = w1[o * 3 + 0] * logit + w1[o * 3 + 1] * wk + w1[o * 3 + 2] * lk + b1[o];
    h1[o] = (s >= 0.0f) ? s : 0.01f * s;
  }
  float h2[8];
  #pragma unroll
  for (int o = 0; o < 8; ++o) {
    float s = b2[o];
    #pragma unroll
    for (int c = 0; c < 32; ++c) s += w2[o * 32 + c] * h1[c];
    h2[o] = (s >= 0.0f) ? s : 0.01f * s;
  }
  float r = b3[0];
  #pragma unroll
  for (int c = 0; c < 8; ++c) r += w3[c] * h2[c];
  return r;
}

// ---------------------------------------------------------------------------
// Kernel 3: fused GEMM (bf16 WMMA) + max/argmax over h + inverse-sigmoid +
// week gather + 3->32->8->1 leaky MLP.  Grid: (ceil(V/256), B), 256 threads.
// Each wave owns 32 v-columns (two 16-wide N-subtiles) x all 64 h-rows.
// ---------------------------------------------------------------------------
__global__ __launch_bounds__(256) void k_main(const uint32_t* __restrict__ Aws,
                                              const uint32_t* __restrict__ embN,
                                              const float* __restrict__ week,
                                              const float* __restrict__ like,
                                              const float* __restrict__ w1, const float* __restrict__ b1,
                                              const float* __restrict__ w2, const float* __restrict__ b2,
                                              const float* __restrict__ w3, const float* __restrict__ b3,
                                              float* __restrict__ out) {
  __shared__ float sA[HP * LDA];   //  9,216 B
  __shared__ float sB[NT * LDB];   // 36,864 B

  const int tid   = threadIdx.x;
  const int lane  = tid & 31;
  const int wave  = tid >> 5;
  const int hi    = lane >> 4;     // lane half per ISA WMMA layouts
  const int nl    = lane & 15;
  const int b     = blockIdx.y;
  const int vBase = blockIdx.x * NT;

  v8f pA0 = {}, pA1 = {}, pA2 = {}, pA3 = {};   // accum, N-subtile 0
  v8f pB0 = {}, pB1 = {}, pB2 = {}, pB3 = {};   // accum, N-subtile 1

  // staging assignments
  const int rowA = tid >> 2, qA = tid & 3;   // 64 rows x (8 dwords/thread)
  const int vB   = vBase + tid;              // one v row per thread (32 dwords)
  const bool vOK = (vB < V);
  const uint32_t* srcA = Aws  + ((size_t)(b * HP + rowA)) * (D / 2);
  const uint32_t* srcB = embN + (size_t)(vOK ? vB : 0) * (D / 2);

  for (int chunk = 0; chunk < D / KC; ++chunk) {
    __syncthreads();
    { // stage A chunk: [64 h][64 k] bf16
      const int s = chunk * (KC / 2) + qA * 8;
      const v4f a0 = *(const v4f*)(srcA + s);
      const v4f a1 = *(const v4f*)(srcA + s + 4);
      *(v4f*)&sA[rowA * LDA + qA * 8]     = a0;
      *(v4f*)&sA[rowA * LDA + qA * 8 + 4] = a1;
    }
    { // stage B chunk: [256 v][64 k] bf16, one row per thread
      const int s = chunk * (KC / 2);
      v4f t0 = {}, t1 = {}, t2 = {}, t3 = {}, t4 = {}, t5 = {}, t6 = {}, t7 = {};
      if (vOK) {
        t0 = *(const v4f*)(srcB + s);      t1 = *(const v4f*)(srcB + s + 4);
        t2 = *(const v4f*)(srcB + s + 8);  t3 = *(const v4f*)(srcB + s + 12);
        t4 = *(const v4f*)(srcB + s + 16); t5 = *(const v4f*)(srcB + s + 20);
        t6 = *(const v4f*)(srcB + s + 24); t7 = *(const v4f*)(srcB + s + 28);
        if (chunk + 1 < D / KC) __builtin_prefetch(srcB + s + 32, 0, 3);
      }
      float* dst = &sB[tid * LDB];
      *(v4f*)(dst)      = t0; *(v4f*)(dst + 4)  = t1;
      *(v4f*)(dst + 8)  = t2; *(v4f*)(dst + 12) = t3;
      *(v4f*)(dst + 16) = t4; *(v4f*)(dst + 20) = t5;
      *(v4f*)(dst + 24) = t6; *(v4f*)(dst + 28) = t7;
    }
    __syncthreads();

    #pragma unroll
    for (int ks = 0; ks < 2; ++ks) {
      // B operands: lane L -> N = L%16, K = 16*hi + 2j (+ ks*32); 8 contiguous dwords
      union { v4f f4[2]; v16bf bf; } b0, b1v;
      const int bb0 = (wave * 32 + nl) * LDB + ks * 16 + 8 * hi;
      const int bb1 = bb0 + 16 * LDB;
      b0.f4[0]  = *(const v4f*)&sB[bb0]; b0.f4[1]  = *(const v4f*)&sB[bb0 + 4];
      b1v.f4[0] = *(const v4f*)&sB[bb1]; b1v.f4[1] = *(const v4f*)&sB[bb1 + 4];
      const int kd = ks * 16 + 8 * hi;
      #pragma unroll
      for (int t = 0; t < 4; ++t) {
        // A operand: lane L -> M = L%16, K = 16*hi + 0..15 (contiguous, matches B)
        union { v4f f4[2]; v16bf bf; } aOp;
        const int ab = (16 * t + nl) * LDA + kd;
        aOp.f4[0] = *(const v4f*)&sA[ab];
        aOp.f4[1] = *(const v4f*)&sA[ab + 4];
        v8f& accA = (t == 0) ? pA0 : (t == 1) ? pA1 : (t == 2) ? pA2 : pA3;
        v8f& accB = (t == 0) ? pB0 : (t == 1) ? pB1 : (t == 2) ? pB2 : pB3;
        accA = __builtin_amdgcn_wmma_f32_16x16x32_bf16(false, aOp.bf, false, b0.bf,
                                                       (short)0, accA, false, false);
        accB = __builtin_amdgcn_wmma_f32_16x16x32_bf16(false, aOp.bf, false, b1v.bf,
                                                       (short)0, accB, false, false);
      }
    }
  }

  // ---- per-column max/argmax over h, then inverse-sigmoid + MLP head
  float best0, best1; int bh0, bh1;
  reduce_cols(pA0, pA1, pA2, pA3, hi, best0, bh0);
  reduce_cols(pB0, pB1, pB2, pB3, hi, best1, bh1);

  const int v0 = vBase + wave * 32 + nl;   // subtile 0 column
  const int v1 = v0 + 16;                  // subtile 1 column
  if (hi == 0) {
    if (v0 < V) {
      const float r = mlp_head(best0, week[b * H + bh0], like[v0], w1, b1, w2, b2, w3, b3);
      out[(size_t)b * V + v0] = r;
    }
    if (v1 < V) {
      const float r = mlp_head(best1, week[b * H + bh1], like[v1], w1, b1, w2, b2, w3, b3);
      out[(size_t)b * V + v1] = r;
    }
  }
}

// ---------------------------------------------------------------------------
extern "C" void kernel_launch(void* const* d_in, const int* in_sizes, int n_in,
                              void* d_out, int out_size, void* d_ws, size_t ws_size,
                              hipStream_t stream) {
  const int*   hist = (const int*)d_in[0];
  const float* week = (const float*)d_in[1];
  const float* emb  = (const float*)d_in[2];
  const float* like = (const float*)d_in[3];
  const float* w1   = (const float*)d_in[4];
  const float* b1   = (const float*)d_in[5];
  const float* w2   = (const float*)d_in[6];
  const float* b2   = (const float*)d_in[7];
  const float* w3   = (const float*)d_in[8];
  const float* b3   = (const float*)d_in[9];
  float* out = (float*)d_out;

  // workspace: [V*256] dwords bf16 normalized table, then [B*64*256] dwords A
  uint32_t* embN = (uint32_t*)d_ws;
  uint32_t* Aws  = embN + (size_t)V * (D / 2);

  k_normalize  <<<V,      128, 0, stream>>>(emb, embN);
  k_gather_hist<<<B * HP, 256, 0, stream>>>(hist, embN, Aws);
  dim3 grid((V + NT - 1) / NT, B);
  k_main<<<grid, 256, 0, stream>>>(Aws, embN, week, like, w1, b1, w2, b2, w3, b3, out);
}
